// StereoDecoder_60842506715307
// MI455X (gfx1250) — matrix operand order repaired and checked by reading, not compiled, not run on hardware
//
#include <hip/hip_runtime.h>
#include <hip/hip_bf16.h>

typedef unsigned short u16;
typedef unsigned int   u32;

typedef __attribute__((ext_vector_type(16))) __bf16 v16bf;
typedef __attribute__((ext_vector_type(8)))  float  v8f;

struct Frag32 { uint4 lo; uint4 hi; };   // 32 bytes = 16 bf16

static __device__ __forceinline__ u16 f2bf(float f) {
    u32 u = __builtin_bit_cast(u32, f);
    u32 r = (u + 0x7FFFu + ((u >> 16) & 1u)) >> 16;   // round-to-nearest-even
    return (u16)r;
}

static __device__ __forceinline__ v16bf load_frag(const u16* p) {
    Frag32 f;
    f.lo = *(const uint4*)(p);
    f.hi = *(const uint4*)(p + 8);
    return __builtin_bit_cast(v16bf, f);
}

// ---------------------------------------------------------------------------
// Problem constants
// ---------------------------------------------------------------------------
#define BB  2
#define CC  32
#define HH  64
#define WW  128
#define DD  48
#define C2  64                   // 2*C input channels of conv
#define DHW (DD*HH*WW)           // 393216 spatial per batch
#define YCST (DD*HH*WW)          // y channel stride (f32 elems)
#define OD 96
#define OH 128
#define OW 256

// ---------------------------------------------------------------------------
// Kernel 1: cost volume directly in bf16, channels-last [b][d][h][w][64]
//   c<32 : l*l + r*r/32,   c>=32 : l*r/32,  r = FR[w+24-d] (zero padded)
// ---------------------------------------------------------------------------
__global__ __launch_bounds__(256)
void build_cost_bf16(const float* __restrict__ FL, const float* __restrict__ FR,
                     u16* __restrict__ cost) {
    int gid = blockIdx.x * 256 + threadIdx.x;          // over B*D*H*W = 786432
    int b = gid / DHW; int rem = gid - b * DHW;
    int d = rem / (HH*WW); rem -= d * (HH*WW);
    int h = rem / WW; int w = rem - h * WW;

    int wr = w + 24 - d;
    bool rv = (unsigned)wr < (unsigned)WW;
    const float* lrow = FL + ((size_t)b * CC * HH + h) * WW + w;
    const float* rrow = FR + ((size_t)b * CC * HH + h) * WW + (rv ? wr : 0);

    u32 pa[16], pb[16];
#pragma unroll
    for (int c = 0; c < 32; ++c) {
        float l = lrow[c * (HH*WW)];
        float r = rv ? rrow[c * (HH*WW)] : 0.f;
        float v1 = l * l + r * r * (1.f / 32.f);
        float v2 = l * r * (1.f / 32.f);
        u32 b1 = f2bf(v1), b2 = f2bf(v2);
        if (c & 1) { pa[c >> 1] |= b1 << 16; pb[c >> 1] |= b2 << 16; }
        else       { pa[c >> 1]  = b1;       pb[c >> 1]  = b2; }
    }
    u32* dst = (u32*)cost + (size_t)gid * 32;          // 64 bf16 = 32 dwords
#pragma unroll
    for (int i = 0; i < 4; ++i) {
        uint4 q; q.x = pa[4*i]; q.y = pa[4*i+1]; q.z = pa[4*i+2]; q.w = pa[4*i+3];
        ((uint4*)dst)[i] = q;
    }
#pragma unroll
    for (int i = 0; i < 4; ++i) {
        uint4 q; q.x = pb[4*i]; q.y = pb[4*i+1]; q.z = pb[4*i+2]; q.w = pb[4*i+3];
        ((uint4*)dst)[4 + i] = q;
    }
}

// ---------------------------------------------------------------------------
// Kernel 2: weights -> bf16 WMMA B-fragments
//   layout: [tap(27)][kchunk(2)][nhalf(2)][lane(32)][16]  (bf16)
//   lane's value s = B[k][n], k = 8*h+s (s<8) or 16+8*h+s-8 (s>=8), h = lane>=16, n = lane&15 + 16*nhalf
// ---------------------------------------------------------------------------
__global__ __launch_bounds__(256)
void prep_weights(const float* __restrict__ w3, u16* __restrict__ wB) {
    int gid = blockIdx.x * 256 + threadIdx.x;          // 1728*32 = 55296
    if (gid >= 1728 * 32) return;
    int n  = gid & 31;
    int k  = gid >> 5;
    int tap = k >> 6;          // 0..26
    int kc  = k & 63;          // input channel 0..63
    int c2  = kc >> 5;         // K chunk of 32
    int kk  = kc & 31;         // row within chunk
    int kd = tap / 9, kh = (tap / 3) % 3, kw = tap % 3;
    int nh   = n >> 4;
    int lane = (n & 15) + (((kk >> 3) & 1) << 4);
    int pos  = (kk & 7) + ((kk >> 4) << 3);
    size_t dst = ((size_t)((tap * 2 + c2) * 2 + nh)) * 512 + (size_t)lane * 16 + pos;
    float v = w3[((((size_t)n * 64 + kc) * 3 + kd) * 3 + kh) * 3 + kw];
    wB[dst] = f2bf(v);
}

// ---------------------------------------------------------------------------
// Kernel 3: implicit-GEMM 3x3x3 conv via v_wmma_f32_16x16x32_bf16, fused BN+ReLU
//   one wave -> 4 M-tiles (64 contiguous w) x 32 out channels; K = 27 taps * 64 ch
//   Per tap/K-chunk: 2 B-frag loads amortized over 4 A tiles -> 16 WMMA / 12 loads
// ---------------------------------------------------------------------------
__global__ __launch_bounds__(256)
void conv3d_bn_relu_wmma(const u16* __restrict__ cost, const u16* __restrict__ wB,
                         const float* __restrict__ gma, const float* __restrict__ bta,
                         const float* __restrict__ mu,  const float* __restrict__ var,
                         float* __restrict__ y) {
    int wid  = threadIdx.x >> 5;
    int lane = threadIdx.x & 31;
    int tile = blockIdx.x * 8 + wid;                   // 12288 tiles
    int b = tile / 6144; int rem = tile - b * 6144;    // 48*64*2 per batch
    int d = rem / 128;   rem -= d * 128;
    int h = rem >> 1;
    int w0 = (rem & 1) << 6;                           // 0 or 64

    int laneLo = lane & 15;
    int hiHalf = lane >> 4;                            // 0 or 1

    // BN constants for this lane's two output channels
    float s0 = gma[laneLo]      * rsqrtf(var[laneLo]      + 1e-5f);
    float s1 = gma[laneLo + 16] * rsqrtf(var[laneLo + 16] + 1e-5f);
    float o0 = bta[laneLo]      - mu[laneLo]      * s0;
    float o1 = bta[laneLo + 16] - mu[laneLo + 16] * s1;

    v8f acc[4][2];
#pragma unroll
    for (int t = 0; t < 4; ++t)
#pragma unroll
        for (int nh = 0; nh < 2; ++nh)
            acc[t][nh] = (v8f){0.f,0.f,0.f,0.f,0.f,0.f,0.f,0.f};
    const uint4 z4 = {0u,0u,0u,0u};

    for (int tap = 0; tap < 27; ++tap) {
        if (tap + 1 < 27) __builtin_prefetch(wB + (size_t)(tap + 1) * 2048, 0, 0);
        int kd = tap / 9, kh = (tap / 3) % 3, kw = tap % 3;
        int di = d + kd - 1, hi_ = h + kh - 1;
        if ((unsigned)di >= (unsigned)DD || (unsigned)hi_ >= (unsigned)HH) continue;

        const u16* rowbase = cost + ((((size_t)b * DD + di) * HH + hi_) * WW) * C2;
        const u16* wtap = wB + (size_t)tap * 2048;
        int wmbase = w0 + laneLo + kw - 1;             // per-lane M (w) position, tile 0

#pragma unroll
        for (int c2 = 0; c2 < 2; ++c2) {
            const u16* wf = wtap + c2 * 1024 + lane * 16;
            v16bf bf0 = load_frag(wf);                 // n 0..15
            v16bf bf1 = load_frag(wf + 512);           // n 16..31
#pragma unroll
            for (int t = 0; t < 4; ++t) {
                int wm = wmbase + 16 * t;
                bool valid = (unsigned)wm < (unsigned)WW;
                Frag32 fa; fa.lo = z4; fa.hi = z4;
                if (valid) {
                    const u16* lp = rowbase + (size_t)wm * C2 + hiHalf * 8 + c2 * 32;
                    fa.lo = *(const uint4*)(lp);       // K {0..7}  (+8 upper lanes)
                    fa.hi = *(const uint4*)(lp + 16);  // K {16..23}(+8 upper lanes)
                }
                v16bf a = __builtin_bit_cast(v16bf, fa);
                acc[t][0] = __builtin_amdgcn_wmma_f32_16x16x32_bf16(
                                false, a, false, bf0, (short)0, acc[t][0], false, false);
                acc[t][1] = __builtin_amdgcn_wmma_f32_16x16x32_bf16(
                                false, a, false, bf1, (short)0, acc[t][1], false, false);
            }
        }
    }

    // C/D layout: VGPR r, lanes0-15 -> (M=r, N=lane); lanes16-31 -> (M=r+8, N=lane-16)
    size_t base0 = ((((size_t)b * CC + laneLo) * DD + d) * HH + h) * WW + w0 + hiHalf * 8;
#pragma unroll
    for (int t = 0; t < 4; ++t) {
#pragma unroll
        for (int nh = 0; nh < 2; ++nh) {
            float s = nh ? s1 : s0;
            float o = nh ? o1 : o0;
            float* yb = y + base0 + (size_t)nh * 16 * YCST + 16 * t;
            float4 q0, q1;
            q0.x = fmaxf(acc[t][nh][0] * s + o, 0.f);
            q0.y = fmaxf(acc[t][nh][1] * s + o, 0.f);
            q0.z = fmaxf(acc[t][nh][2] * s + o, 0.f);
            q0.w = fmaxf(acc[t][nh][3] * s + o, 0.f);
            q1.x = fmaxf(acc[t][nh][4] * s + o, 0.f);
            q1.y = fmaxf(acc[t][nh][5] * s + o, 0.f);
            q1.z = fmaxf(acc[t][nh][6] * s + o, 0.f);
            q1.w = fmaxf(acc[t][nh][7] * s + o, 0.f);
            *(float4*)(yb)     = q0;
            *(float4*)(yb + 4) = q1;
        }
    }
}

// ---------------------------------------------------------------------------
// Kernel 4: fused stride-2 transposed conv + softmax over depth (96)
//   out[b,od,oh,ow] = sum_c sum_{i,j,k} y[b,c,i,j,k] * wt[c,0,od-2i+1,oh-2j+1,ow-2k+1]
//   then softmax(-out) over od. One block per (b,oh,ow), 128 threads (od<96 active).
// ---------------------------------------------------------------------------
static __device__ __forceinline__ int taps1d(int o, int lim, int* idx, int* wsel) {
    int n = 0;
    if (o & 1) {
        int i0 = (o + 1) >> 1;
        if (i0 < lim) { idx[n] = i0; wsel[n] = 0; ++n; }
        idx[n] = (o - 1) >> 1; wsel[n] = 2; ++n;
    } else {
        idx[0] = o >> 1; wsel[0] = 1; n = 1;
    }
    return n;
}

__global__ __launch_bounds__(128)
void tconv_softmax(const float* __restrict__ y, const float* __restrict__ wt,
                   const float* __restrict__ bt, float* __restrict__ out) {
    __shared__ float swt[32 * 27];
    __shared__ float red[128];
    int tid = threadIdx.x;
    int ow = blockIdx.x, oh = blockIdx.y, b = blockIdx.z;

    for (int i = tid; i < 32 * 27; i += 128) swt[i] = wt[i];   // wt[c][0][kd][kh][kw]
    __syncthreads();

    int od = tid;
    float val = 0.f;
    if (od < OD) {
        int ii[2], ia[2], jj[2], ja[2], kk[2], ka[2];
        int ni = taps1d(od, DD, ii, ia);
        int nj = taps1d(oh, HH, jj, ja);
        int nk = taps1d(ow, WW, kk, ka);
        for (int di = 0; di < ni; ++di)
            for (int dj = 0; dj < nj; ++dj)
                for (int dk = 0; dk < nk; ++dk) {
                    int t = ia[di] * 9 + ja[dj] * 3 + ka[dk];
                    const float* yb = y + ((size_t)b * CC * DD + ii[di]) * (HH*WW)
                                        + jj[dj] * WW + kk[dk];
                    float s = 0.f;
#pragma unroll
                    for (int c = 0; c < 32; ++c)
                        s += yb[(size_t)c * YCST] * swt[c * 27 + t];
                    val += s;
                }
    }
    float v = -(val + bt[0]);
    red[tid] = (od < OD) ? v : -3.4e38f;
    __syncthreads();
    for (int st = 64; st > 0; st >>= 1) {
        if (tid < st) red[tid] = fmaxf(red[tid], red[tid + st]);
        __syncthreads();
    }
    float mx = red[0];
    __syncthreads();
    float e = (od < OD) ? __expf(v - mx) : 0.f;
    red[tid] = e;
    __syncthreads();
    for (int st = 64; st > 0; st >>= 1) {
        if (tid < st) red[tid] += red[tid + st];
        __syncthreads();
    }
    float inv = 1.f / red[0];
    if (od < OD)
        out[(((size_t)b * OD + od) * OH + oh) * OW + ow] = e * inv;
}

// ---------------------------------------------------------------------------
// Launch
// ---------------------------------------------------------------------------
extern "C" void kernel_launch(void* const* d_in, const int* in_sizes, int n_in,
                              void* d_out, int out_size, void* d_ws, size_t ws_size,
                              hipStream_t stream) {
    const float* FL  = (const float*)d_in[0];
    const float* FR  = (const float*)d_in[1];
    const float* w3  = (const float*)d_in[2];
    const float* gma = (const float*)d_in[3];
    const float* bta = (const float*)d_in[4];
    const float* mu  = (const float*)d_in[5];
    const float* var = (const float*)d_in[6];
    const float* wt  = (const float*)d_in[7];
    const float* bt  = (const float*)d_in[8];
    float* outp = (float*)d_out;

    char* ws = (char*)d_ws;
    const size_t costBytes = (size_t)BB * DD * HH * WW * C2 * sizeof(u16);   // 100,663,296
    const size_t wBBytes   = (size_t)27 * 2 * 2 * 32 * 16 * sizeof(u16);     //     110,592
    u16*  cost = (u16*)ws;
    u16*  wB   = (u16*)(ws + costBytes);
    float* y   = (float*)(ws + costBytes + wBBytes);

    build_cost_bf16<<<(BB * DHW) / 256, 256, 0, stream>>>(FL, FR, cost);
    prep_weights<<<(1728 * 32 + 255) / 256, 256, 0, stream>>>(w3, wB);
    conv3d_bn_relu_wmma<<<12288 / 8, 256, 0, stream>>>(cost, wB, gma, bta, mu, var, y);
    tconv_softmax<<<dim3(OW, OH, BB), 128, 0, stream>>>(y, wt, bt, outp);
}
